// CGAE_18528488915637
// MI455X (gfx1250) — compile-verified, matrix operand order checked by reading, not run on hardware
//
#include <hip/hip_runtime.h>
#include <hip/hip_bf16.h>

typedef __bf16 bhalf;
typedef __attribute__((ext_vector_type(16))) __bf16 v16bf;
typedef __attribute__((ext_vector_type(8)))  __bf16 v8bf;
typedef __attribute__((ext_vector_type(8)))  float  v8f;
typedef __attribute__((ext_vector_type(4)))  float  v4f;
typedef unsigned int u32x4 __attribute__((ext_vector_type(4)));
typedef int          i32x8 __attribute__((ext_vector_type(8)));
typedef int          i32x4 __attribute__((ext_vector_type(4)));

#define NROWS 16384
#define NFEAT 128
#define NHID  64
#define NOUT  128

static constexpr size_t Z_ORI  = 0;
static constexpr size_t Z_AUG  = (size_t)NROWS * NHID;
static constexpr size_t XH_ORI = 2 * (size_t)NROWS * NHID;
static constexpr size_t XH_AUG = XH_ORI + (size_t)NROWS * NOUT;

// -------------------------------------------------------------------------
// Pass 0: support1 = {feat, feat_a} @ W1  -> bf16, TRANSPOSED [128][NROWS]
// -------------------------------------------------------------------------
__global__ __launch_bounds__(128) void support1_kernel(
    const float* __restrict__ feat, const float* __restrict__ feat_a,
    const float* __restrict__ W1, bhalf* __restrict__ Bt1)
{
    __shared__ float w[NFEAT * NHID];              // 32 KB
    const int tid = threadIdx.x;
    for (int i = tid; i < NFEAT * NHID; i += 128) w[i] = W1[i];
    __syncthreads();

    const int strm = blockIdx.y;
    const float* x = strm ? feat_a : feat;
    const int r = blockIdx.x * 128 + tid;
    const float* xr = x + (size_t)r * NFEAT;

    float acc[NHID] = {};
    for (int f = 0; f < NFEAT; ++f) {
        const float a = xr[f];
        #pragma unroll
        for (int h = 0; h < NHID; ++h) acc[h] = fmaf(a, w[f * NHID + h], acc[h]);
    }
    bhalf* bt = Bt1 + (size_t)strm * NHID * NROWS;
    #pragma unroll
    for (int h = 0; h < NHID; ++h) bt[(size_t)h * NROWS + r] = (bhalf)acc[h];
}

// -------------------------------------------------------------------------
// Pass 2: support2 = z @ W2 -> bf16, TRANSPOSED [256][NROWS]
// -------------------------------------------------------------------------
__global__ __launch_bounds__(128) void support2_kernel(
    const float* __restrict__ out, const float* __restrict__ W2,
    bhalf* __restrict__ Bt2)
{
    __shared__ float w[NHID * NOUT];               // 32 KB
    const int tid = threadIdx.x;
    for (int i = tid; i < NHID * NOUT; i += 128) w[i] = W2[i];
    __syncthreads();

    const int strm  = blockIdx.y >> 1;
    const int ohalf = blockIdx.y & 1;
    const float* z = out + (strm ? Z_AUG : Z_ORI);
    const int r = blockIdx.x * 128 + tid;
    const float* zr = z + (size_t)r * NHID;

    float acc[64] = {};
    for (int h = 0; h < NHID; ++h) {
        const float a = zr[h];
        #pragma unroll
        for (int j = 0; j < 64; ++j)
            acc[j] = fmaf(a, w[h * NOUT + ohalf * 64 + j], acc[j]);
    }
    bhalf* bt = Bt2 + (size_t)strm * NOUT * NROWS;
    #pragma unroll
    for (int j = 0; j < 64; ++j)
        bt[(size_t)(ohalf * 64 + j) * NROWS + r] = (bhalf)acc[j];
}

// -------------------------------------------------------------------------
// TDM issue: DMA a [NCOLS rows x 64 K] bf16 tile (row stride NROWS elements)
// from global Bt into LDS, with hardware padding: after every 128B of data,
// insert 16B -> LDS row stride 144B (36 banks: conflict-free ds_load_b128).
// D# bitfields per CDNA5 ISA 8.3/8.4 (2-D tile; groups 2/3 zero => <=2D).
// -------------------------------------------------------------------------
__device__ __forceinline__ void tdm_issue_b(const bhalf* gsrc, unsigned lds_off,
                                            int ncols)
{
    const unsigned long long ga = (unsigned long long)(size_t)gsrc;
    u32x4 g0;
    g0[0] = 1u;                                     // count=1, user desc
    g0[1] = lds_off;                                // lds_addr (bytes)
    g0[2] = (unsigned)ga;                           // global_addr[31:0]
    g0[3] = (unsigned)(ga >> 32) | (2u << 30);      // global_addr[56:32] | type=2
    i32x8 g1;
    g1[0] = (int)((1u << 16)      // data_size = 1 (2 bytes)
                | (1u << 20)      // pad_enable
                | (4u << 22)      // pad_interval = 4 -> every 32 DWORDs (128B)
                | (3u << 25));    // pad_amount   = 3 -> 4 DWORDs (16B)
    g1[1] = (int)(((unsigned)NROWS & 0xFFFFu) << 16);   // tensor_dim0[15:0]
    g1[2] = (int)(((unsigned)NROWS >> 16) | ((unsigned)ncols << 16)); // dim0 hi | dim1 lo
    g1[3] = (int)(64u << 16);                           // tile_dim0 = 64
    g1[4] = (int)(unsigned)ncols;                       // tile_dim1 = ncols, tile_dim2 = 0
    g1[5] = (int)NROWS;                                 // tensor_dim0_stride
    g1[6] = 0;
    g1[7] = 0;
    const i32x4 z4 = {0, 0, 0, 0};
    const i32x8 z8 = {0, 0, 0, 0, 0, 0, 0, 0};
    __builtin_amdgcn_tensor_load_to_lds(g0, g1, z4, z4, z8, 0);
}

// -------------------------------------------------------------------------
// Big GEMM: C[NROWS, NCOLS] = fadj(f32) @ B, B supplied bf16-transposed as
// Bt[NCOLS][NROWS]. fadj converted f32->bf16 in registers (HBM stream).
// B 64-K chunks staged block-wide into LDS by the Tensor Data Mover (double-
// buffered, TENSORcnt + barrier synced); all 8 waves read via ds_load_b128.
// Per barrier round: 2 K-substeps x 8 WMMA. B fragments are rotated through
// a 2-deep pipeline: frag(t+1) ds_loads issue before the WMMA on frag(t).
// -------------------------------------------------------------------------
template <int NCOLS, int PASS>
__global__ __launch_bounds__(256) void adj_gemm_kernel(
    const float* __restrict__ A, const bhalf* __restrict__ Bt,
    float* __restrict__ out)
{
    constexpr int WAVES_N = NCOLS / 128;   // 1 (pass1) or 2 (pass3)
    constexpr int WAVES_M = 8 / WAVES_N;   // 8 or 4
    constexpr int ROW_E   = 72;            // staged LDS row: 64 bf16 + 8 pad = 144B

    __shared__ __align__(16) bhalf btile[2][NCOLS * ROW_E];   // 36KB / 72KB

    const int tid  = threadIdx.x;
    const int lane = tid & 31;
    const int wave = tid >> 5;
    const int mWave = blockIdx.x * (16 * WAVES_M) + (wave / WAVES_N) * 16;
    const int nBase = (wave % WAVES_N) * 128;

    const int row = lane & 15;             // M row (A) / N col (B) within tile
    const int kg  = (lane >> 4) * 8;       // K sub-group per lane half

    v8f acc[8] = {};

    const float* aRow = A + (size_t)(mWave + row) * NROWS + kg;
    const unsigned ldsBase = (unsigned)(size_t)(void*)&btile[0][0];
    constexpr unsigned BUF_BYTES = (unsigned)NCOLS * ROW_E * sizeof(bhalf);

    if (wave == 0) tdm_issue_b(Bt, ldsBase, NCOLS);          // prologue: buf 0

    for (int k0 = 0; k0 < NROWS; k0 += 64) {
        const int cur = (k0 >> 6) & 1;
        if (wave == 0) __builtin_amdgcn_s_wait_tensorcnt(0); // buf[cur] landed
        __syncthreads();                                     // visible to all;
                                                             // prev reads of buf[cur^1] done
        if (wave == 0 && k0 + 64 < NROWS)
            tdm_issue_b(Bt + (size_t)(k0 + 64),
                        ldsBase + (unsigned)(cur ^ 1) * BUF_BYTES, NCOLS);

        const bhalf* bBase = &btile[cur][(unsigned)(nBase + row) * ROW_E + kg];

        #pragma unroll
        for (int s = 0; s < 2; ++s) {
            const int ks = k0 + 32 * s;
            // ---- A fragment: 16 fp32 -> bf16 (K: kg+0..7, kg+16..23) ----
            __builtin_prefetch(aRow + ks + 256, 0, 1);       // global_prefetch_b8
            v4f a0 = *(const v4f*)(aRow + ks);
            v4f a1 = *(const v4f*)(aRow + ks + 4);
            v4f a2 = *(const v4f*)(aRow + ks + 16);
            v4f a3 = *(const v4f*)(aRow + ks + 20);
            v16bf af;
            #pragma unroll
            for (int i = 0; i < 4; ++i) {
                af[i]      = (bhalf)a0[i];
                af[4 + i]  = (bhalf)a1[i];
                af[8 + i]  = (bhalf)a2[i];
                af[12 + i] = (bhalf)a3[i];
            }
            // ---- B fragments: 2-deep rotation so ds_load(t+1) overlaps wmma(t)
            const bhalf* bp0 = bBase + 32 * s;
            v8bf c0 = *(const v8bf*)(bp0);
            v8bf c1 = *(const v8bf*)(bp0 + 16);
            v16bf bcur = __builtin_shufflevector(
                c0, c1, 0, 1, 2, 3, 4, 5, 6, 7, 8, 9, 10, 11, 12, 13, 14, 15);
            #pragma unroll
            for (int t = 0; t < 8; ++t) {
                v16bf bnext = bcur;
                if (t < 7) {
                    const bhalf* bp = bBase + (unsigned)(t + 1) * (16 * ROW_E) + 32 * s;
                    v8bf b0 = *(const v8bf*)(bp);         // K = kg+0..7
                    v8bf b1 = *(const v8bf*)(bp + 16);    // K = kg+16..23
                    bnext = __builtin_shufflevector(
                        b0, b1, 0, 1, 2, 3, 4, 5, 6, 7, 8, 9, 10, 11, 12, 13, 14, 15);
                }
                acc[t] = __builtin_amdgcn_wmma_f32_16x16x32_bf16(
                    false, af, false, bcur, (short)0, acc[t], false, false);
                bcur = bnext;
            }
        }
    }

    // ---- Epilogue: C layout lane0-15 -> M=vgpr, lane16-31 -> M=vgpr+8 ----
    #pragma unroll
    for (int t = 0; t < 8; ++t) {
        const int n = nBase + 16 * t + (lane & 15);
        #pragma unroll
        for (int r = 0; r < 8; ++r) {
            const int m = mWave + r + ((lane >> 4) << 3);
            const float v = acc[t][r];
            size_t idx;
            if (PASS == 1) {
                idx = (n < NHID) ? (Z_ORI + (size_t)m * NHID + n)
                                 : (Z_AUG + (size_t)m * NHID + (n - NHID));
            } else {
                idx = (n < NOUT) ? (XH_ORI + (size_t)m * NOUT + n)
                                 : (XH_AUG + (size_t)m * NOUT + (n - NOUT));
            }
            out[idx] = v;
        }
    }
}

extern "C" void kernel_launch(void* const* d_in, const int* in_sizes, int n_in,
                              void* d_out, int out_size, void* d_ws, size_t ws_size,
                              hipStream_t stream) {
    const float* feat   = (const float*)d_in[0];
    const float* feat_a = (const float*)d_in[1];
    const float* fadj   = (const float*)d_in[2];
    const float* W1     = (const float*)d_in[3];
    const float* W2     = (const float*)d_in[4];
    float* out = (float*)d_out;

    bhalf* Bt1 = (bhalf*)d_ws;                                      // 4 MB
    bhalf* Bt2 = (bhalf*)((char*)d_ws +
                          (size_t)2 * NHID * NROWS * sizeof(bhalf)); // next 8 MB

    // Pass 0: support1 (both streams) -> bf16 transposed
    support1_kernel<<<dim3(NROWS / 128, 2), 128, 0, stream>>>(feat, feat_a, W1, Bt1);
    // Pass 1: z = fadj @ support1, fused N=128 (one fadj sweep for both streams)
    adj_gemm_kernel<128, 1><<<dim3(NROWS / 128), 256, 0, stream>>>(fadj, Bt1, out);
    // Pass 2: support2 = z @ W2 (both streams, both column halves)
    support2_kernel<<<dim3(NROWS / 128, 4), 128, 0, stream>>>(out, W2, Bt2);
    // Pass 3: xhat = fadj @ support2, fused N=256 (one fadj sweep)
    adj_gemm_kernel<256, 3><<<dim3(NROWS / 64), 256, 0, stream>>>(fadj, Bt2, out);
}
